// ForwardOnlyCPNN_46918222741664
// MI455X (gfx1250) — matrix operand order repaired and checked by reading, not compiled, not run on hardware
//
#include <hip/hip_runtime.h>
#include <hip/hip_bf16.h>
#include <stdint.h>

#define BATCH 8192
#define IN    1024
#define HID   4096
#define OUT   1000
#define NCHUNK 4
#define CHUNK_H (HID / NCHUNK)   // 1024

#define TM 128
#define TN 128
#define TK 32
#define KT (IN / TK)        // 32 k-steps
#define NT (CHUNK_H / TN)   // 8 n-tiles per chunk

typedef __attribute__((ext_vector_type(16))) __bf16 v16bf;
typedef __attribute__((ext_vector_type(8)))  __bf16 v8bf;
typedef __attribute__((ext_vector_type(8)))  float  v8f;

union Frag16 { v16bf v; v8bf h[2]; };

static __device__ __forceinline__ unsigned short f32_to_bf16_rne(float f) {
    union { float f; unsigned int u; } c; c.f = f;
    unsigned int u = c.u;
    unsigned int r;
    if ((u & 0x7FFFFFFFu) > 0x7F800000u) r = u | 0x00400000u;      // quiet NaN
    else r = u + 0x7FFFu + ((u >> 16) & 1u);                        // RNE
    return (unsigned short)(r >> 16);
}

// 16-byte async DMA: global -> LDS, tracked by ASYNCcnt (no VGPR staging).
// Byte offset is a compile-time immediate (24-bit signed IOFFSET field).
// No "memory" clobber: ordering vs. consumers is enforced by wait_async0()
// (volatile + memory) and __syncthreads(); volatile asm stays mutually ordered.
#define ASYNC_LD16(ldsOff, gBase, immOff)                                   \
    asm volatile("global_load_async_to_lds_b128 %0, %1, off offset:%c2"     \
                 :: "v"(ldsOff), "v"(gBase), "i"(immOff))

static __device__ __forceinline__ void wait_async0() {
    asm volatile("s_wait_asynccnt 0x0" ::: "memory");
}

// ---------------- Kernel 1: f32 -> bf16 bulk convert (float4 granular) ----
__global__ void cvt_bf16_kernel(const float* __restrict__ src,
                                unsigned short* __restrict__ dst, int n4) {
    int i = blockIdx.x * blockDim.x + threadIdx.x;
    if (i >= n4) return;
    float4 f = ((const float4*)src)[i];
    ushort4 o;
    o.x = f32_to_bf16_rne(f.x);
    o.y = f32_to_bf16_rne(f.y);
    o.z = f32_to_bf16_rne(f.z);
    o.w = f32_to_bf16_rne(f.w);
    ((ushort4*)dst)[i] = o;
}

// ---------------- Kernel 2: w2[h] = sum_k W[h,k]^2 (f32) ------------------
__global__ void rowsumsq_kernel(const float* __restrict__ w, float* __restrict__ w2) {
    __shared__ float red[256];
    const int h = blockIdx.x;
    const float* row = w + (size_t)h * IN;
    float s = 0.f;
    for (int k = threadIdx.x; k < IN; k += 256) { float v = row[k]; s += v * v; }
    red[threadIdx.x] = s; __syncthreads();
    for (int off = 128; off > 0; off >>= 1) {
        if (threadIdx.x < off) red[threadIdx.x] += red[threadIdx.x + off];
        __syncthreads();
    }
    if (threadIdx.x == 0) w2[h] = red[0];
}

// ---------------- Kernel 3: fused bf16 WMMA GEMM + running argmin ---------
__global__ __launch_bounds__(256, 1)
void gemm_argmin_kernel(const unsigned short* __restrict__ xb,   // [BATCH][IN] bf16
                        const unsigned short* __restrict__ wb,   // [HID][IN]  bf16
                        const float* __restrict__ w2,            // [HID]
                        float* __restrict__ pval,                // [NCHUNK][BATCH]
                        int* __restrict__ pidx) {                // [NCHUNK][BATCH]
    __shared__ unsigned short At[2][TM * TK];   // 8 KB per stage
    __shared__ unsigned short Bt[2][TM * TK];   // 8 KB per stage
    __shared__ float redV[TM * 4];
    __shared__ int   redI[TM * 4];

    const int tid    = threadIdx.x;
    const int lane   = tid & 31;
    const int wave   = tid >> 5;
    const int waveM  = wave & 1;     // 2 waves along M
    const int waveN  = wave >> 1;    // 4 waves along N
    const int laneLo = lane & 15;
    const int laneHi = lane >> 4;

    const int mBase = blockIdx.x * TM;          // batch-row base
    const int hBase = blockIdx.y * CHUNK_H;     // hidden base of this chunk

    // running per-lane argmin: rows covered = waveM*64 + mi*16 + laneHi*8 + r
    float minv[4][8];
    int   mini[4][8];
#pragma unroll
    for (int mi = 0; mi < 4; ++mi)
#pragma unroll
        for (int r = 0; r < 8; ++r) { minv[mi][r] = 3.4e38f; mini[mi][r] = 0x7fffffff; }

    // 512 x 16B segments per tile, 2 per thread; thread t owns segs t, t+256
    const int segRow0 = (tid      ) >> 2, segSg0 = (tid      ) & 3;
    const int segRow1 = (tid + 256) >> 2, segSg1 = (tid + 256) & 3;

    // hoisted global base pointers (k-step advance goes into the imm offset)
    const uint4* xg = (const uint4*)xb;   // row stride IN/8 uint4
    const uint4* wg = (const uint4*)wb;
    const uint4* aP0 = &xg[(size_t)(mBase + segRow0) * (IN / 8) + segSg0];
    const uint4* aP1 = &xg[(size_t)(mBase + segRow1) * (IN / 8) + segSg1];

    // hoisted LDS destination offsets (2 stages x {A,B} x 2 segs)
    unsigned ldsA[2][2], ldsB[2][2];
#pragma unroll
    for (int s = 0; s < 2; ++s) {
        ldsA[s][0] = (unsigned)(uintptr_t)&At[s][segRow0 * TK + segSg0 * 8];
        ldsA[s][1] = (unsigned)(uintptr_t)&At[s][segRow1 * TK + segSg1 * 8];
        ldsB[s][0] = (unsigned)(uintptr_t)&Bt[s][segRow0 * TK + segSg0 * 8];
        ldsB[s][1] = (unsigned)(uintptr_t)&Bt[s][segRow1 * TK + segSg1 * 8];
    }

    for (int nt = 0; nt < NT; ++nt) {
        const int nTileBase = hBase + nt * TN;
        const uint4* bP0 = &wg[(size_t)(nTileBase + segRow0) * (IN / 8) + segSg0];
        const uint4* bP1 = &wg[(size_t)(nTileBase + segRow1) * (IN / 8) + segSg1];

        v8f acc[4][2];
        const v8f vzero = {0.f, 0.f, 0.f, 0.f, 0.f, 0.f, 0.f, 0.f};
#pragma unroll
        for (int mi = 0; mi < 4; ++mi)
#pragma unroll
            for (int ni = 0; ni < 2; ++ni) acc[mi][ni] = vzero;

        // compute one 32-wide k-step from LDS buffer `cur` (cur is literal)
        auto compute_step = [&](int cur) {
            // A fragments: ISA 16-bit A 16x32 layout
            // lane<16 : elems 0..7 = K0..7,  elems 8..15 = K16..23
            // lane>=16: elems 0..7 = K8..15, elems 8..15 = K24..31
            Frag16 afr[4];
#pragma unroll
            for (int mi = 0; mi < 4; ++mi) {
                const int row = waveM * 64 + mi * 16 + laneLo;
                const unsigned short* p = &At[cur][row * TK + laneHi * 8];
                afr[mi].h[0] = *(const v8bf*)p;
                afr[mi].h[1] = *(const v8bf*)(p + 16);
            }
            // B fragments: N = lane&15, same K layout as A (lane-half h: K 8h..8h+7, 16+8h..16+8h+7)
            Frag16 bfr[2];
#pragma unroll
            for (int ni = 0; ni < 2; ++ni) {
                const int row = waveN * 32 + ni * 16 + laneLo;
                const unsigned short* p = &Bt[cur][row * TK + laneHi * 8];
                bfr[ni].h[0] = *(const v8bf*)p;
                bfr[ni].h[1] = *(const v8bf*)(p + 16);
            }
#pragma unroll
            for (int mi = 0; mi < 4; ++mi)
#pragma unroll
                for (int ni = 0; ni < 2; ++ni)
                    acc[mi][ni] = __builtin_amdgcn_wmma_f32_16x16x32_bf16(
                        false, afr[mi].v, false, bfr[ni].v,
                        (short)0, acc[mi][ni], false, false);
        };

        __syncthreads();      // previous n-tile finished reading both buffers
        ASYNC_LD16(ldsA[0][0], aP0, 0);
        ASYNC_LD16(ldsB[0][0], bP0, 0);
        ASYNC_LD16(ldsA[0][1], aP1, 0);
        ASYNC_LD16(ldsB[0][1], bP1, 0);

        // fully unrolled steady state: no loop-carried phi on the accumulators
#pragma unroll
        for (int kt = 0; kt < KT - 1; ++kt) {
            wait_async0();    // my stage-kt DMAs have landed in LDS
            __syncthreads();  // everyone's landed; prev readers of buf kt+1 done
            {                 // prefetch stage kt+1 (overlaps with compute below)
                const int s = (kt + 1) & 1;
                ASYNC_LD16(ldsA[s][0], aP0, (kt + 1) * (TK * 2));
                ASYNC_LD16(ldsB[s][0], bP0, (kt + 1) * (TK * 2));
                ASYNC_LD16(ldsA[s][1], aP1, (kt + 1) * (TK * 2));
                ASYNC_LD16(ldsB[s][1], bP1, (kt + 1) * (TK * 2));
            }
            compute_step(kt & 1);
        }
        // epilogue: final k-step, nothing left to prefetch
        wait_async0();
        __syncthreads();
        compute_step((KT - 1) & 1);

        // --- fold this n-tile into running argmin: d = ||w||^2 - 2*s
#pragma unroll
        for (int ni = 0; ni < 2; ++ni) {
            const int ncol = nTileBase + waveN * 32 + ni * 16 + laneLo;
            const float wn = w2[ncol];
#pragma unroll
            for (int mi = 0; mi < 4; ++mi)
#pragma unroll
                for (int r = 0; r < 8; ++r) {
                    float d = wn - 2.0f * acc[mi][ni][r];
                    if (d < minv[mi][r] || (d == minv[mi][r] && ncol < mini[mi][r])) {
                        minv[mi][r] = d; mini[mi][r] = ncol;
                    }
                }
        }
    }

    // --- cross-lane (16-wide butterfly) then cross-wave-N reduce via LDS
    __syncthreads();
#pragma unroll
    for (int mi = 0; mi < 4; ++mi)
#pragma unroll
        for (int r = 0; r < 8; ++r) {
            float v = minv[mi][r]; int ix = mini[mi][r];
#pragma unroll
            for (int m = 1; m < 16; m <<= 1) {
                float ov = __shfl_xor(v, m, 16);
                int   oi = __shfl_xor(ix, m, 16);
                if (ov < v || (ov == v && oi < ix)) { v = ov; ix = oi; }
            }
            if (laneLo == 0) {
                const int row = waveM * 64 + mi * 16 + laneHi * 8 + r;
                redV[row * 4 + waveN] = v;
                redI[row * 4 + waveN] = ix;
            }
        }
    __syncthreads();
    if (tid < TM) {
        float v = redV[tid * 4]; int ix = redI[tid * 4];
#pragma unroll
        for (int j = 1; j < 4; ++j) {
            float ov = redV[tid * 4 + j]; int oi = redI[tid * 4 + j];
            if (ov < v || (ov == v && oi < ix)) { v = ov; ix = oi; }
        }
        const int g = mBase + tid;
        pval[blockIdx.y * BATCH + g] = v;
        pidx[blockIdx.y * BATCH + g] = ix;
    }
}

// ---------------- Kernel 4: reduce chunks -> winner; gather output --------
__global__ void finalize_kernel(const float* __restrict__ pval, const int* __restrict__ pidx,
                                const float* __restrict__ gw, float* __restrict__ out,
                                int* __restrict__ winners, int* __restrict__ count) {
    __shared__ int sWin;
    const int b = blockIdx.x;
    if (threadIdx.x == 0) {
        float v = pval[b]; int ix = pidx[b];
        for (int c = 1; c < NCHUNK; ++c) {
            float ov = pval[c * BATCH + b]; int oi = pidx[c * BATCH + b];
            if (ov < v || (ov == v && oi < ix)) { v = ov; ix = oi; }
        }
        sWin = ix;
        winners[b] = ix;
        if (b == 0) *count = BATCH;
    }
    __syncthreads();
    const int win = sWin;
    for (int o = threadIdx.x; o < OUT; o += blockDim.x)
        out[(size_t)b * OUT + o] = gw[(size_t)o * HID + win];
}

// --------------------------------------------------------------------------
extern "C" void kernel_launch(void* const* d_in, const int* in_sizes, int n_in,
                              void* d_out, int out_size, void* d_ws, size_t ws_size,
                              hipStream_t stream) {
    const float* x  = (const float*)d_in[0];   // [BATCH, IN]
    const float* kw = (const float*)d_in[1];   // [HID, IN]
    const float* gw = (const float*)d_in[2];   // [OUT, HID]

    char* ws = (char*)d_ws;
    unsigned short* xb = (unsigned short*)ws;                                   // 16 MB
    unsigned short* wb = (unsigned short*)(ws + (size_t)BATCH * IN * 2);        //  8 MB
    float* w2   = (float*)(ws + (size_t)BATCH * IN * 2 + (size_t)HID * IN * 2); // 16 KB
    float* pval = (float*)((char*)w2 + (size_t)HID * sizeof(float));            // 128 KB
    int*   pidx = (int*)((char*)pval + (size_t)NCHUNK * BATCH * sizeof(float)); // 128 KB

    cvt_bf16_kernel<<<(BATCH * IN / 4 + 255) / 256, 256, 0, stream>>>(x, xb, BATCH * IN / 4);
    cvt_bf16_kernel<<<(HID * IN / 4 + 255) / 256, 256, 0, stream>>>(kw, wb, HID * IN / 4);
    rowsumsq_kernel<<<HID, 256, 0, stream>>>(kw, w2);

    dim3 g(BATCH / TM, NCHUNK);
    gemm_argmin_kernel<<<g, 256, 0, stream>>>(xb, wb, w2, pval, pidx);

    float* out     = (float*)d_out;
    int*   winners = (int*)(out + (size_t)BATCH * OUT);
    int*   count   = winners + BATCH;
    finalize_kernel<<<BATCH, 256, 0, stream>>>(pval, pidx, gw, out, winners, count);
}
